// PartRefinement_72756745994442
// MI455X (gfx1250) — matrix operand order repaired and checked by reading, not compiled, not run on hardware
//
#include <hip/hip_runtime.h>

// ---------------------------------------------------------------------------
// MI455X (gfx1250) implementation.
// All heavy contractions go through v_wmma_f32_16x16x32_bf16 (f32 accumulate).
// Pipeline:
//   k_gather  : projection + masked gather  -> P   (B,960,N) bf16
//   k_gemm fc : IP[b,c,o] = sum_n P*fc_w^T  -> IP  (B,960,N) bf16  (+fc_b over o)
//   k_gemm cb : codebias[o,b] from code / global_code (K=1024 each)
//   k_gemm c1 : f1 = c1_w[:,2053:]*IP + c1_b        (B,1024,N) bf16
//   k_c1x     : f1 += codebias1+codebias2 + grid/level0 terms
//   k_gemm c2,c3(relu), w2d1(sum_k), w2d2, w2d3(x2 strided calls, +f64 residual),
//          c4, c5, c6(f32 out)
//   k_permute : (B,6,1024) -> (B,2048,3)
// GEMM kernel: 256 thr = 8 waves, 64M x 64N block, each wave owns two 16x16
// tiles sharing one A fragment (2 wmma per k-step).  LDS rows padded to 40
// bf16 (80B = 20 dwords) so the 16 fragment rows hit 16 distinct banks while
// every 16B chunk stays 16B aligned.
// ---------------------------------------------------------------------------

typedef __bf16 bf16_t;
typedef bf16_t v16bf __attribute__((ext_vector_type(16)));
typedef bf16_t v8bf  __attribute__((ext_vector_type(8)));
typedef bf16_t v4bf  __attribute__((ext_vector_type(4)));
typedef float  v8f   __attribute__((ext_vector_type(8)));
typedef float  v4f   __attribute__((ext_vector_type(4)));

union ABf { v16bf v; v8bf h[2]; };
union P8  { v8bf v; bf16_t e[8]; };
union P4  { v4bf v; bf16_t e[4]; };

#define B_  32
#define N_  1024

// ---------------- projection gather ----------------
__global__ __launch_bounds__(256) void k_gather(
    const float* __restrict__ level0,
    const float* __restrict__ img0, const float* __restrict__ img1,
    const float* __restrict__ img2, const float* __restrict__ img3,
    bf16_t* __restrict__ P)
{
  int b  = blockIdx.y;
  int n0 = blockIdx.x * 16;
  int t  = threadIdx.x;
  __shared__ float sh[16], sw[16];
  if (t < 16) {
    const float* L = level0 + (size_t)b * 3 * N_;
    int n = n0 + t;
    float X = L[n], Y = L[N_ + n], Z = L[2 * N_ + n];
    float h = 248.0f * (Y / Z) + 111.5f;
    float w = 248.0f * (X / (-Z)) + 111.5f;
    sh[t] = fminf(fmaxf(h, 0.0f), 223.0f);
    sw[t] = fminf(fmaxf(w, 0.0f), 223.0f);
  }
  __syncthreads();
  for (int e = t; e < 960 * 16; e += 256) {
    int c_all = e >> 4, nl = e & 15;
    const float* fb; int s; float invd;
    if (c_all < 64)       { s = 56; invd = 4.0f;  fb = img0 + (size_t)b*64*3136  + (size_t)c_all      *3136; }
    else if (c_all < 192) { s = 28; invd = 8.0f;  fb = img1 + (size_t)b*128*784  + (size_t)(c_all-64) *784;  }
    else if (c_all < 448) { s = 14; invd = 16.0f; fb = img2 + (size_t)b*256*196  + (size_t)(c_all-192)*196;  }
    else                  { s = 7;  invd = 32.0f; fb = img3 + (size_t)b*512*49   + (size_t)(c_all-448)*49;   }
    float x = sh[nl] / invd;     // 224/S is an exact power of two for all levels
    float y = sw[nl] / invd;
    int x1 = (int)floorf(x);
    int x2 = (int)fminf(ceilf(x), (float)(s - 1));
    int y1 = (int)floorf(y);
    int y2 = (int)fminf(ceilf(y), (float)(s - 1));
    int xi = (int)x, yi = (int)y;
    float w11 = (float)((x2 - xi) * (y2 - yi));
    float w12 = (float)((x2 - xi) * (yi - y1));
    float w21 = (float)((xi - x1) * (y2 - yi));
    float w22 = (float)((xi - x1) * (yi - y1));
    float q11 = fb[x1 * s + y1], q12 = fb[x1 * s + y2];
    float q21 = fb[x2 * s + y1], q22 = fb[x2 * s + y2];
    float v = w11 * q11 + w21 * q21 + w12 * q12 + w22 * q22;
    P[(size_t)b * 960 * N_ + (size_t)c_all * N_ + n0 + nl] = (bf16_t)v;
  }
}

// ---------------- generic batched GEMM via WMMA ----------------
// D[b,m,n] = sum_k A[b][m][k] * X[b][k][n]  (+biasM[m] +biasN[n] +add)  (relu?)
// AM: 0 = f32 row-major, unaligned rows (scalar loads)
//     1 = f32 row-major, 16B-aligned rows (vec4 loads)
//     2 = bf16 row-major, 16B-aligned rows (vec8 loads)
// XM: 0 = f32, xsK==1 (k-contiguous, vec4 loads) with n guard
//     1 = bf16, xsN==1 (n-contiguous, vec8 loads); requires Nn % 64 == 0
template <int AM, int XM>
__global__ __launch_bounds__(256) void k_gemm(
    const void* __restrict__ Aptr, long long aBatch, int ldA,
    const void* __restrict__ Xptr, long long xBatch, long long xsK, long long xsN,
    const float* __restrict__ biasM, const float* __restrict__ biasN,
    const bf16_t* __restrict__ addPtr, long long addBatch, long long addRow,
    void* __restrict__ outPtr, int outF32, long long outBatch, long long outRow,
    int M, int Nn, int K, int relu)
{
  __shared__ __align__(16) bf16_t As[64][40];   // [m][k], 80B row stride
  __shared__ __align__(16) bf16_t Bs[64][40];   // [n][k]
  int t = threadIdx.x, lane = t & 31, wv = t >> 5;
  int wm = wv & 3, wn = wv >> 2;                // 4x2 waves, each wave 16M x 32N
  int bz = blockIdx.z;
  int m0 = blockIdx.y * 64, n0 = blockIdx.x * 64;
  long long ab = (long long)bz * aBatch;
  long long xb = (long long)bz * xBatch;
  v8f acc0 = {}, acc1 = {};

  for (int k0 = 0; k0 < K; k0 += 32) {
    // ---- stage A tile (64m x 32k) ----
    if (AM == 2) {
      int r = t >> 2, kc = (t & 3) * 8;         // 256 items, 1 iter
      P8 pk; 
      if (m0 + r < M) {
        pk.v = *(const v8bf*)((const bf16_t*)Aptr + ab + (long long)(m0 + r) * ldA + k0 + kc);
      } else {
        #pragma unroll
        for (int j = 0; j < 8; ++j) pk.e[j] = (bf16_t)0.0f;
      }
      *(v8bf*)&As[r][kc] = pk.v;
    } else {
      #pragma unroll
      for (int i = t; i < 512; i += 256) {      // 2 iters
        int r = i >> 3, kk = (i & 7) * 4;
        P4 pk;
        if (m0 + r < M) {
          const float* ap = (const float*)Aptr + ab + (long long)(m0 + r) * ldA + k0 + kk;
          if (AM == 1) {
            v4f f = *(const v4f*)ap;
            #pragma unroll
            for (int j = 0; j < 4; ++j) pk.e[j] = (bf16_t)f[j];
          } else {
            #pragma unroll
            for (int j = 0; j < 4; ++j) pk.e[j] = (bf16_t)ap[j];
          }
        } else {
          #pragma unroll
          for (int j = 0; j < 4; ++j) pk.e[j] = (bf16_t)0.0f;
        }
        *(v4bf*)&As[r][kk] = pk.v;
      }
    }
    // ---- stage X tile (64n x 32k) into Bs[n][k] ----
    if (XM == 1) {
      // lane -> k column, so a wave's 32 lanes write 2B-adjacent cols of one row
      int kk = t & 31, nc = (t >> 5) * 8;       // 256 items, 1 iter
      P8 pk;
      pk.v = *(const v8bf*)((const bf16_t*)Xptr + xb + (long long)(k0 + kk) * xsK + (n0 + nc));
      #pragma unroll
      for (int j = 0; j < 8; ++j) Bs[nc + j][kk] = pk.e[j];
    } else {
      #pragma unroll
      for (int i = t; i < 512; i += 256) {      // 2 iters
        int nn = i >> 3, kk = (i & 7) * 4;
        P4 pk;
        if (n0 + nn < Nn) {
          const float* xp = (const float*)Xptr + xb + (long long)(k0 + kk) + (long long)(n0 + nn) * xsN;
          v4f f = *(const v4f*)xp;              // xsK == 1 for XM==0
          #pragma unroll
          for (int j = 0; j < 4; ++j) pk.e[j] = (bf16_t)f[j];
        } else {
          #pragma unroll
          for (int j = 0; j < 4; ++j) pk.e[j] = (bf16_t)0.0f;
        }
        *(v4bf*)&Bs[nn][kk] = pk.v;
      }
    }
    __syncthreads();
    // A layout: lane<16 -> M=lane, K=0..15 ; lane>=16 -> M=lane-16, K=16..31
    ABf a;
    int arow = wm * 16 + (lane & 15);
    int kb = (lane >> 4) * 16;
    a.h[0] = *(const v8bf*)&As[arow][kb];
    a.h[1] = *(const v8bf*)&As[arow][kb + 8];
    // B layout (contiguous k, same as A): lane<16 -> N=lane, K=0..15 ; lane>=16 -> N=lane-16, K=16..31
    ABf b0, b1;
    int brow = wn * 32 + (lane & 15);
    int kb2 = (lane >> 4) * 16;
    b0.h[0] = *(const v8bf*)&Bs[brow][kb2];
    b0.h[1] = *(const v8bf*)&Bs[brow][kb2 + 8];
    b1.h[0] = *(const v8bf*)&Bs[brow + 16][kb2];
    b1.h[1] = *(const v8bf*)&Bs[brow + 16][kb2 + 8];
    acc0 = __builtin_amdgcn_wmma_f32_16x16x32_bf16(false, a.v, false, b0.v,
                                                   (short)0, acc0, false, false);
    acc1 = __builtin_amdgcn_wmma_f32_16x16x32_bf16(false, a.v, false, b1.v,
                                                   (short)0, acc1, false, false);
    __syncthreads();
  }

  // store: VGPR r -> M = r (lanes 0-15) / r+8 (lanes 16-31); lane%16 -> N
  int ngB = n0 + wn * 32 + (lane & 15);
  int mb = m0 + wm * 16 + ((lane >> 4) << 3);
  long long ob = (long long)bz * outBatch;
  long long db = (long long)bz * addBatch;
  #pragma unroll
  for (int sub = 0; sub < 2; ++sub) {
    int ng = ngB + sub * 16;
    #pragma unroll
    for (int r = 0; r < 8; ++r) {
      int mg = mb + r;
      if (mg < M && ng < Nn) {
        float v = sub ? acc1[r] : acc0[r];
        if (biasM)  v += biasM[mg];
        if (biasN)  v += biasN[ng];
        if (addPtr) v += (float)addPtr[db + (long long)mg * addRow + ng];
        if (relu)   v = fmaxf(v, 0.0f);
        long long oi = ob + (long long)mg * outRow + ng;
        if (outF32) ((float*)outPtr)[oi] = v;
        else        ((bf16_t*)outPtr)[oi] = (bf16_t)v;
      }
    }
  }
}

// ---------------- c1 epilogue: code/grid/level0 channels ----------------
__global__ __launch_bounds__(256) void k_c1x(
    bf16_t* __restrict__ f1, const float* __restrict__ cb1,
    const float* __restrict__ cb2, const float* __restrict__ c1w,
    const float* __restrict__ level0)
{
  size_t idx = (size_t)blockIdx.x * 256 + threadIdx.x;   // B*1024*1024
  if (idx >= (size_t)B_ * 1024 * N_) return;
  int n = idx & (N_ - 1);
  size_t r = idx >> 10;
  int o = (int)(r & 1023);
  int b = (int)(r >> 10);
  const float* wr = c1w + (size_t)o * 3013;
  const float* L  = level0 + (size_t)b * 3 * N_;
  float g0 = (n & 1) ? 0.2f : -0.2f;
  float extra = cb1[o * 32 + b] + cb2[o * 32 + b]
              + wr[0] * g0 + wr[1] * (-0.2f)
              + wr[2] * L[n] + wr[3] * L[N_ + n] + wr[4] * L[2 * N_ + n];
  f1[idx] = (bf16_t)((float)f1[idx] + extra);
}

__global__ void k_w1sum(const float* __restrict__ w2d1, float* __restrict__ w1s) {
  int i = blockIdx.x * 256 + threadIdx.x;
  if (i < 64 * 64) w1s[i] = w2d1[2 * i] + w2d1[2 * i + 1];
}

__global__ void k_permute(const float* __restrict__ c6o, float* __restrict__ out) {
  int idx = blockIdx.x * 256 + threadIdx.x;          // B*2048*3
  if (idx >= B_ * 2048 * 3) return;
  int c = idx % 3;
  int r = idx / 3;
  int m = r & 2047;
  int b = r >> 11;
  int ch = 2 * c + (m >> 10);
  out[idx] = c6o[(size_t)b * 6144 + (size_t)ch * N_ + (m & (N_ - 1))];
}

// ---------------- workspace layout (bytes) : peak ~277 MB ----------------
#define OFF_P      0ull                         // (B,960,N) bf16  62,914,560
#define OFF_IP     62914560ull                  // (B,960,N) bf16
#define OFF_F1     125829120ull                 // (B,1024,N) bf16 67,108,864
#define OFF_F128   192937984ull                 // (B,128,N) bf16
#define OFF_F64    201326592ull                 // (B,64,N) bf16
#define OFF_F2A    205520896ull                 // (B,64,N) bf16
#define OFF_F2B    209715200ull                 // (B,128,N) bf16
#define OFF_FSUM   218103808ull                 // (B,64,N) bf16
#define OFF_G512A  222298112ull                 // (B,512,N) bf16 33,554,432
#define OFF_G512B  255852544ull                 // (B,512,N) bf16
#define OFF_CB1    289406976ull                 // (1024,32) f32
#define OFF_CB2    289538048ull
#define OFF_W1S    289669120ull                 // (64,64) f32
#define OFF_C6     289685504ull                 // (B,6,N) f32

extern "C" void kernel_launch(void* const* d_in, const int* in_sizes, int n_in,
                              void* d_out, int out_size, void* d_ws, size_t ws_size,
                              hipStream_t stream) {
  const float* level0 = (const float*)d_in[0];
  const float* code   = (const float*)d_in[1];
  const float* gcode  = (const float*)d_in[2];
  const float* img0   = (const float*)d_in[3];
  const float* img1   = (const float*)d_in[4];
  const float* img2   = (const float*)d_in[5];
  const float* img3   = (const float*)d_in[6];
  const float* fc_w = (const float*)d_in[7];  const float* fc_b = (const float*)d_in[8];
  const float* c1_w = (const float*)d_in[9];  const float* c1_b = (const float*)d_in[10];
  const float* c2_w = (const float*)d_in[11]; const float* c2_b = (const float*)d_in[12];
  const float* c3_w = (const float*)d_in[13]; const float* c3_b = (const float*)d_in[14];
  const float* w2d1 = (const float*)d_in[15]; const float* b2d1 = (const float*)d_in[16];
  const float* w2d2 = (const float*)d_in[17]; const float* b2d2 = (const float*)d_in[18];
  const float* w2d3 = (const float*)d_in[19]; const float* b2d3 = (const float*)d_in[20];
  const float* c4_w = (const float*)d_in[21]; const float* c4_b = (const float*)d_in[22];
  const float* c5_w = (const float*)d_in[23]; const float* c5_b = (const float*)d_in[24];
  const float* c6_w = (const float*)d_in[25]; const float* c6_b = (const float*)d_in[26];

  char* ws = (char*)d_ws;
  bf16_t* P     = (bf16_t*)(ws + OFF_P);
  bf16_t* IP    = (bf16_t*)(ws + OFF_IP);
  bf16_t* f1    = (bf16_t*)(ws + OFF_F1);
  bf16_t* f128  = (bf16_t*)(ws + OFF_F128);
  bf16_t* f64   = (bf16_t*)(ws + OFF_F64);
  bf16_t* f2a   = (bf16_t*)(ws + OFF_F2A);
  bf16_t* f2b   = (bf16_t*)(ws + OFF_F2B);
  bf16_t* fsum  = (bf16_t*)(ws + OFF_FSUM);
  bf16_t* g512a = (bf16_t*)(ws + OFF_G512A);
  bf16_t* g512b = (bf16_t*)(ws + OFF_G512B);
  float*  cb1   = (float*)(ws + OFF_CB1);
  float*  cb2   = (float*)(ws + OFF_CB2);
  float*  w1s   = (float*)(ws + OFF_W1S);
  float*  c6o   = (float*)(ws + OFF_C6);

  // 1. projection gather
  k_gather<<<dim3(N_ / 16, B_), 256, 0, stream>>>(level0, img0, img1, img2, img3, P);

  // 2. fc over points: IP[b,c,o] = sum_n P[b,c,n] fc_w[o,n] + fc_b[o]
  k_gemm<2, 0><<<dim3(16, 15, B_), 256, 0, stream>>>(
      P, 960LL * N_, N_, fc_w, 0, 1, N_, nullptr, fc_b,
      nullptr, 0, 0, IP, 0, 960LL * N_, N_, 960, N_, N_, 0);

  // 3. code biases: cb[o,b] = sum_c c1_w[o,5+c]*code[b,c]
  k_gemm<0, 0><<<dim3(1, 16, 1), 256, 0, stream>>>(
      c1_w + 5, 0, 3013, code, 0, 1, 1024, nullptr, nullptr,
      nullptr, 0, 0, cb1, 1, 0, 32, 1024, 32, 1024, 0);
  k_gemm<0, 0><<<dim3(1, 16, 1), 256, 0, stream>>>(
      c1_w + 1029, 0, 3013, gcode, 0, 1, 1024, nullptr, nullptr,
      nullptr, 0, 0, cb2, 1, 0, 32, 1024, 32, 1024, 0);

  // 4. c1 main GEMM over the 960 image channels + epilogue
  k_gemm<0, 1><<<dim3(16, 16, B_), 256, 0, stream>>>(
      c1_w + 2053, 0, 3013, IP, 960LL * N_, N_, 1, c1_b, nullptr,
      nullptr, 0, 0, f1, 0, 1024LL * N_, N_, 1024, N_, 960, 0);
  k_c1x<<<(B_ * 1024 * N_) / 256, 256, 0, stream>>>(f1, cb1, cb2, c1_w, level0);

  // 5. c2, c3(relu)
  k_gemm<1, 1><<<dim3(16, 2, B_), 256, 0, stream>>>(
      c2_w, 0, 1024, f1, 1024LL * N_, N_, 1, c2_b, nullptr,
      nullptr, 0, 0, f128, 0, 128LL * N_, N_, 128, N_, 1024, 0);
  k_gemm<1, 1><<<dim3(16, 1, B_), 256, 0, stream>>>(
      c3_w, 0, 128, f128, 128LL * N_, N_, 1, c3_b, nullptr,
      nullptr, 0, 0, f64, 0, 64LL * N_, N_, 64, N_, 128, 1);

  // 6. f2 chain
  k_w1sum<<<16, 256, 0, stream>>>(w2d1, w1s);
  k_gemm<1, 1><<<dim3(16, 1, B_), 256, 0, stream>>>(
      w1s, 0, 64, f64, 64LL * N_, N_, 1, b2d1, nullptr,
      nullptr, 0, 0, f2a, 0, 64LL * N_, N_, 64, N_, 64, 0);
  k_gemm<1, 1><<<dim3(16, 2, B_), 256, 0, stream>>>(
      w2d2, 0, 64, f2a, 64LL * N_, N_, 1, b2d2, nullptr,
      nullptr, 0, 0, f2b, 0, 128LL * N_, N_, 128, N_, 64, 0);
  // w2d3 on the (u,64) split, fused residual add of f64 -> fsum
  for (int u = 0; u < 2; ++u) {
    k_gemm<1, 1><<<dim3(16, 1, B_), 256, 0, stream>>>(
        w2d3, 0, 64, f2b + (size_t)u * 64 * N_, 128LL * N_, N_, 1,
        b2d3, nullptr, f64 + (size_t)u * N_, 64LL * N_, 2LL * N_,
        fsum + (size_t)u * N_, 0, 64LL * N_, 2LL * N_, 32, N_, 64, 0);
  }

  // 7. c4, c5, c6
  k_gemm<1, 1><<<dim3(16, 8, B_), 256, 0, stream>>>(
      c4_w, 0, 64, fsum, 64LL * N_, N_, 1, c4_b, nullptr,
      nullptr, 0, 0, g512a, 0, 512LL * N_, N_, 512, N_, 64, 0);
  k_gemm<1, 1><<<dim3(16, 8, B_), 256, 0, stream>>>(
      c5_w, 0, 512, g512a, 512LL * N_, N_, 1, c5_b, nullptr,
      nullptr, 0, 0, g512b, 0, 512LL * N_, N_, 512, N_, 512, 0);
  k_gemm<1, 1><<<dim3(16, 1, B_), 256, 0, stream>>>(
      c6_w, 0, 512, g512b, 512LL * N_, N_, 1, c6_b, nullptr,
      nullptr, 0, 0, c6o, 1, 6LL * N_, N_, 6, N_, 512, 0);

  // 8. (B,6,1024) -> (B,2048,3)
  k_permute<<<(B_ * 2048 * 3) / 256, 256, 0, stream>>>(c6o, (float*)d_out);
}
